// GeometricMambaLayer_56736517980360
// MI455X (gfx1250) — matrix operand order repaired and checked by reading, not compiled, not run on hardware
//
#include <hip/hip_runtime.h>
#include <math.h>

typedef __attribute__((ext_vector_type(16))) _Float16 v16h;
typedef __attribute__((ext_vector_type(8)))  _Float16 v8h;
typedef __attribute__((ext_vector_type(8)))  float    v8f;

#define BB   4
#define TT   2048
#define DD   1024
#define DIN  2048
#define ROWS (BB*TT)   // 8192

#define BN 64          // block tile N
#define BK 32          // K step
#define LDSB_STRIDE 40 // halves per LDS row (32 data + 8 pad -> 80B, 16B aligned, bank-spread)

// Detect gfx1250 async Global->LDS path.
#if defined(__has_builtin)
#if __has_builtin(__builtin_amdgcn_global_load_async_to_lds_b128) && \
    __has_builtin(__builtin_amdgcn_s_wait_asynccnt)
#define USE_ASYNC_LDS 1
#endif
#endif
#ifndef USE_ASYNC_LDS
#define USE_ASYNC_LDS 0
#endif

#if USE_ASYNC_LDS
// Builtin signature (from hipcc diagnostic): (v4i as1*, v4i as3*, imm int, imm int)
typedef int v4i_b __attribute__((vector_size(16)));
typedef __attribute__((address_space(1))) v4i_b* gv4i_p;
typedef __attribute__((address_space(3))) v4i_b* lv4i_p;
#endif

// ---------------- WMMA fragment loads ----------------
// A: M x K row-major f16.  Documented 16-bit A layout (16x32):
//   lanes 0-15 : M=lane,    elems 0-7 = K 0..7,  elems 8-15 = K 16..23
//   lanes 16-31: M=lane-16, elems 0-7 = K 8..15, elems 8-15 = K 24..31
__device__ __forceinline__ v16h load_afrag(const _Float16* __restrict__ A, int lda,
                                           int m0, int kb, int lane) {
  int m  = m0 + (lane & 15);
  int ko = kb + ((lane >> 4) << 3);
  const _Float16* p = A + (size_t)m * lda + ko;
  v8h lo = *(const v8h*)(p);
  v8h hi = *(const v8h*)(p + 16);
  v16h r;
#pragma unroll
  for (int i = 0; i < 8; ++i) { r[i] = lo[i]; r[i + 8] = hi[i]; }
  return r;
}

// B fragment from the LDS-staged tile (row n_local holds K=kb..kb+31 contiguous).
// 16-bit B layout (matches A): lanes 0-15 hold K 0..7,16..23; lanes 16-31 hold K 8..15,24..31,
// N = lane & 15.
__device__ __forceinline__ v16h load_bfrag_lds(const _Float16* base, int n0, int lane) {
  int n  = n0 + (lane & 15);
  int ko = (lane >> 4) << 3;
  const _Float16* p = base + n * LDSB_STRIDE + ko;
  v8h lo = *(const v8h*)(p);
  v8h hi = *(const v8h*)(p + 16);
  v16h r;
#pragma unroll
  for (int i = 0; i < 8; ++i) { r[i] = lo[i]; r[i + 8] = hi[i]; }
  return r;
}

// Stage one 64(N) x 32(K) f16 tile of Bt (N x K row-major) into LDS.
// 256 threads x 16B = 4KB. ASYNCcnt-tracked async copy when available.
__device__ __forceinline__ void stageB(const _Float16* __restrict__ Bt, int K, int nB,
                                       int kb, _Float16* dst, int tid) {
  int row = tid >> 2;
  int col = (tid & 3) << 3;
  const _Float16* src = Bt + (size_t)(nB + row) * K + kb + col;
  _Float16* d = dst + row * LDSB_STRIDE + col;
#if USE_ASYNC_LDS
  __builtin_amdgcn_global_load_async_to_lds_b128((gv4i_p)src, (lv4i_p)d, 0, 0);
#else
  *(v8h*)d = *(const v8h*)src;
#endif
}

// ---------------- epilogue ----------------
// C/D layout: lane holds N = n_tile + (lane&15); VGPR r holds M = m_tile + 8*(lane>>4) + r
// MODE 0: out_h = f16(acc + bias)            (x_inner)
// MODE 1: out_f = acc + bias                 (pre-activation)
// MODE 2: out_h = f16((inv@W_sp+b_sp) * silu(acc + bias))   (fused gate)
template <int MODE>
__device__ __forceinline__ void epilogue(v8f c, int mt, int nt, int lane, int N,
                                         const float* __restrict__ bias,
                                         float* __restrict__ outF,
                                         _Float16* __restrict__ outH,
                                         const float* __restrict__ invb,
                                         const float* __restrict__ Wsp,
                                         const float* __restrict__ bsp) {
  int n  = nt + (lane & 15);
  int mb = mt + ((lane >> 4) << 3);
#pragma unroll
  for (int r = 0; r < 8; ++r) {
    int m = mb + r;
    float acc = c[r] + bias[n];
    if (MODE == 0) {
      outH[(size_t)m * N + n] = (_Float16)acc;
    } else if (MODE == 1) {
      outF[(size_t)m * N + n] = acc;
    } else {
      float g = acc / (1.0f + __expf(-acc));   // silu
      float y = bsp[n];
#pragma unroll
      for (int j = 0; j < 8; ++j) y += invb[m * 8 + j] * Wsp[j * N + n];
      outH[(size_t)m * N + n] = (_Float16)(y * g);
    }
  }
}

// ---------------- tiled WMMA GEMM ----------------
// block = 256 threads = 8 waves (4 in M x 2 in N); wave tile 32x32 (2x2 WMMA),
// block tile 128x64. B tile double-buffered in LDS via async Global->LDS copies.
template <int MODE>
__global__ __launch_bounds__(256) void gemm_wmma(
    const _Float16* __restrict__ A, const _Float16* __restrict__ Bt,
    const float* __restrict__ bias, float* __restrict__ outF,
    _Float16* __restrict__ outH, const float* __restrict__ invb,
    const float* __restrict__ Wsp, const float* __restrict__ bsp,
    int M, int N, int K) {
  (void)M;
  __shared__ _Float16 ldsB[2][BN * LDSB_STRIDE];

  const int tid  = threadIdx.x;
  const int lane = tid & 31;
  const int w    = tid >> 5;
  const int m0   = blockIdx.x * 128 + (w & 3) * 32;
  const int nB   = blockIdx.y * BN;        // block column base (global)
  const int n0   = (w >> 2) * 32;          // wave column offset within block

  stageB(Bt, K, nB, 0, ldsB[0], tid);      // prologue stage of first K slice

  v8f c00 = {}, c01 = {}, c10 = {}, c11 = {};
  int buf = 0;
  for (int kb = 0; kb < K; kb += BK, buf ^= 1) {
#if USE_ASYNC_LDS
    __builtin_amdgcn_s_wait_asynccnt(0);   // my staging writes for ldsB[buf] done
#endif
    __syncthreads();                       // everyone staged buf / done reading buf^1
    if (kb + BK < K) {
      stageB(Bt, K, nB, kb + BK, ldsB[buf ^ 1], tid);
      // uniform-condition prefetch of the next A slice
      __builtin_prefetch(A + (size_t)(m0 + (lane & 15)) * K + kb + BK, 0, 1);
    }
    v16h a0 = load_afrag(A, K, m0, kb, lane);
    v16h a1 = load_afrag(A, K, m0 + 16, kb, lane);
    v16h b0 = load_bfrag_lds(ldsB[buf], n0, lane);
    v16h b1 = load_bfrag_lds(ldsB[buf], n0 + 16, lane);
    c00 = __builtin_amdgcn_wmma_f32_16x16x32_f16(false, a0, false, b0, (short)0, c00, false, false);
    c01 = __builtin_amdgcn_wmma_f32_16x16x32_f16(false, a0, false, b1, (short)0, c01, false, false);
    c10 = __builtin_amdgcn_wmma_f32_16x16x32_f16(false, a1, false, b0, (short)0, c10, false, false);
    c11 = __builtin_amdgcn_wmma_f32_16x16x32_f16(false, a1, false, b1, (short)0, c11, false, false);
  }
  epilogue<MODE>(c00, m0,      nB + n0,      lane, N, bias, outF, outH, invb, Wsp, bsp);
  epilogue<MODE>(c01, m0,      nB + n0 + 16, lane, N, bias, outF, outH, invb, Wsp, bsp);
  epilogue<MODE>(c10, m0 + 16, nB + n0,      lane, N, bias, outF, outH, invb, Wsp, bsp);
  epilogue<MODE>(c11, m0 + 16, nB + n0 + 16, lane, N, bias, outF, outH, invb, Wsp, bsp);
}

// ---------------- conversion kernels ----------------
__global__ void k_cvt_f16(const float* __restrict__ src, _Float16* __restrict__ dst, int n) {
  int i = blockIdx.x * 256 + threadIdx.x;
  if (i < n) dst[i] = (_Float16)src[i];
}

// W: K x N row-major f32  ->  Wt: N x K row-major f16
__global__ void k_cvt_trans(const float* __restrict__ W, _Float16* __restrict__ Wt,
                            int K, int N) {
  int o = blockIdx.x * 256 + threadIdx.x;
  if (o >= K * N) return;
  int n = o / K, k = o - n * K;
  Wt[o] = (_Float16)W[(size_t)k * N + n];
}

// ---------------- basis projection + 16-wide LayerNorm ----------------
// one wave per row; 8 rows per block
__global__ __launch_bounds__(256) void k_wri(const _Float16* __restrict__ xin,
                                             const float* __restrict__ br,
                                             const float* __restrict__ bi,
                                             const float* __restrict__ lnw,
                                             const float* __restrict__ lnb,
                                             float* __restrict__ vec) {
  int lane = threadIdx.x & 31;
  int row  = blockIdx.x * 8 + (threadIdx.x >> 5);
  const _Float16* xr = xin + (size_t)row * DIN;
  float acc[16];
#pragma unroll
  for (int j = 0; j < 16; ++j) acc[j] = 0.0f;
  for (int k = lane; k < DIN; k += 32) {
    float xv = (float)xr[k];
    const float* brp = br + k * 8;
    const float* bip = bi + k * 8;
#pragma unroll
    for (int j = 0; j < 8; ++j) {
      acc[j]     += xv * brp[j];
      acc[8 + j] += xv * bip[j];
    }
  }
#pragma unroll
  for (int off = 16; off > 0; off >>= 1)
#pragma unroll
    for (int j = 0; j < 16; ++j) acc[j] += __shfl_xor(acc[j], off, 32);
  if (lane == 0) {
    float m = 0.0f;
#pragma unroll
    for (int j = 0; j < 16; ++j) m += acc[j];
    m *= (1.0f / 16.0f);
    float va = 0.0f;
#pragma unroll
    for (int j = 0; j < 16; ++j) { float d = acc[j] - m; va += d * d; }
    va *= (1.0f / 16.0f);
    float is = rsqrtf(va + 1e-5f);
#pragma unroll
    for (int j = 0; j < 16; ++j)
      vec[(size_t)row * 16 + j] = (acc[j] - m) * is * lnw[j] + lnb[j];
  }
}

// ---------------- sequential scan (4 threads, one per batch) ----------------
__global__ void k_scan(const float* __restrict__ vec, const float* __restrict__ theta,
                       const float* __restrict__ omega, const float* __restrict__ sigma,
                       float* __restrict__ invb, float* __restrict__ fstate) {
  int b = threadIdx.x;
  if (b >= BB) return;
  float ct[8], st[8], sc[8], hr[8], hi[8];
#pragma unroll
  for (int j = 0; j < 8; ++j) {
    ct[j] = cosf(theta[j]);
    st[j] = sinf(theta[j]);
    float s = fminf(fmaxf(sigma[j], 0.5f), 2.0f);
    sc[j] = s * (1.0f + 0.1f * omega[j]);
    hr[j] = 0.0f; hi[j] = 0.0f;
  }
  const float SIGN[8] = {1.f, -1.f, -1.f, -1.f, -1.f, -1.f, -1.f, -1.f};
  for (int t = 0; t < TT; ++t) {
    size_t rb = (size_t)b * TT + t;
    const float* v = vec + rb * 16;
    float cr[8], ci[8], rr[8], ri[8], nr[8], ni[8];
#pragma unroll
    for (int j = 0; j < 8; ++j) { cr[j] = v[j]; ci[j] = v[8 + j]; }
#pragma unroll
    for (int j = 0; j < 8; ++j) {
      rr[j] = (hr[j] * ct[j] - hi[j] * st[j]) * sc[j];
      ri[j] = (hr[j] * st[j] + hi[j] * ct[j]) * sc[j];
    }
#pragma unroll
    for (int j = 0; j < 4; ++j) {
      nr[j]     = rr[j] * cr[j]     - cr[j + 4] * rr[j + 4];
      ni[j]     = cr[j + 4] * rr[j] + rr[j + 4] * cr[j];
      nr[j + 4] = ri[j] * ci[j]     - ci[j + 4] * ri[j + 4];
      ni[j + 4] = ci[j + 4] * ri[j] + ri[j + 4] * ci[j];
    }
#pragma unroll
    for (int j = 0; j < 8; ++j) {
      hr[j] = nr[j]; hi[j] = ni[j];
      invb[rb * 8 + j] = SIGN[j] * (nr[j] * nr[j] + ni[j] * ni[j]);
    }
  }
#pragma unroll
  for (int j = 0; j < 8; ++j) {
    fstate[b * 16 + j]     = hr[j];
    fstate[b * 16 + 8 + j] = hi[j];
  }
}

// ---------------- final residual LayerNorm over D=1024 ----------------
__global__ __launch_bounds__(256) void k_final_ln(const float* __restrict__ x,
                                                  const float* __restrict__ pre,
                                                  const float* __restrict__ nw,
                                                  const float* __restrict__ nb,
                                                  float* __restrict__ out) {
  __shared__ float sw[8];
  __shared__ float sstat;
  int row = blockIdx.x;
  int tid = threadIdx.x, lane = tid & 31, w = tid >> 5;
  const float* xr = x + (size_t)row * DD;
  const float* pr = pre + (size_t)row * DD;
  float v[4];
  float s = 0.0f;
#pragma unroll
  for (int i = 0; i < 4; ++i) {
    v[i] = xr[tid + 256 * i] + pr[tid + 256 * i];
    s += v[i];
  }
#pragma unroll
  for (int off = 16; off > 0; off >>= 1) s += __shfl_xor(s, off, 32);
  if (lane == 0) sw[w] = s;
  __syncthreads();
  if (tid == 0) {
    float t = 0.0f;
    for (int i = 0; i < 8; ++i) t += sw[i];
    sstat = t * (1.0f / DD);
  }
  __syncthreads();
  float mean = sstat;
  __syncthreads();
  s = 0.0f;
#pragma unroll
  for (int i = 0; i < 4; ++i) { float d = v[i] - mean; s += d * d; }
#pragma unroll
  for (int off = 16; off > 0; off >>= 1) s += __shfl_xor(s, off, 32);
  if (lane == 0) sw[w] = s;
  __syncthreads();
  if (tid == 0) {
    float t = 0.0f;
    for (int i = 0; i < 8; ++i) t += sw[i];
    sstat = rsqrtf(t * (1.0f / DD) + 1e-5f);
  }
  __syncthreads();
  float isd = sstat;
#pragma unroll
  for (int i = 0; i < 4; ++i) {
    int c = tid + 256 * i;
    out[(size_t)row * DD + c] = (v[i] - mean) * isd * nw[c] + nb[c];
  }
}

// ---------------- host launcher ----------------
extern "C" void kernel_launch(void* const* d_in, const int* in_sizes, int n_in,
                              void* d_out, int out_size, void* d_ws, size_t ws_size,
                              hipStream_t stream) {
  (void)in_sizes; (void)n_in; (void)out_size; (void)ws_size;
  const float* x     = (const float*)d_in[0];
  const float* W_in  = (const float*)d_in[1];
  const float* b_in  = (const float*)d_in[2];
  const float* br    = (const float*)d_in[3];
  const float* bi    = (const float*)d_in[4];
  const float* lnw   = (const float*)d_in[5];
  const float* lnb   = (const float*)d_in[6];
  const float* theta = (const float*)d_in[7];
  const float* omega = (const float*)d_in[8];
  const float* sigma = (const float*)d_in[9];
  const float* W_sp  = (const float*)d_in[10];
  const float* b_sp  = (const float*)d_in[11];
  const float* W_g   = (const float*)d_in[12];
  const float* b_g   = (const float*)d_in[13];
  const float* W_out = (const float*)d_in[14];
  const float* b_out = (const float*)d_in[15];
  const float* nw    = (const float*)d_in[16];
  const float* nb    = (const float*)d_in[17];
  float* out = (float*)d_out;

  char* ws = (char*)d_ws;
  size_t off = 0;
  auto alloc = [&](size_t bytes) -> char* {
    char* p = ws + off;
    off += (bytes + 255) & ~(size_t)255;
    return p;
  };
  _Float16* xh    = (_Float16*)alloc((size_t)ROWS * DD * 2);    // 16 MB
  _Float16* WinT  = (_Float16*)alloc((size_t)DIN * DD * 2);     //  4 MB
  _Float16* WgT   = (_Float16*)alloc((size_t)DIN * DIN * 2);    //  8 MB
  _Float16* WoutT = (_Float16*)alloc((size_t)DD * DIN * 2);     //  4 MB
  _Float16* xinH  = (_Float16*)alloc((size_t)ROWS * DIN * 2);   // 32 MB
  float*    vec   = (float*)   alloc((size_t)ROWS * 16 * 4);    // 0.5 MB
  float*    invb  = (float*)   alloc((size_t)ROWS * 8 * 4);     // 0.25 MB
  _Float16* h16   = (_Float16*)alloc((size_t)ROWS * DIN * 2);   // 32 MB
  float*    pre   = (float*)   alloc((size_t)ROWS * DD * 4);    // 32 MB

  // one-time precision conversion / weight transposes
  k_cvt_f16<<<(ROWS * DD) / 256, 256, 0, stream>>>(x, xh, ROWS * DD);
  k_cvt_trans<<<(DD * DIN) / 256, 256, 0, stream>>>(W_in, WinT, DD, DIN);
  k_cvt_trans<<<(DIN * DIN) / 256, 256, 0, stream>>>(W_g, WgT, DIN, DIN);
  k_cvt_trans<<<(DIN * DD) / 256, 256, 0, stream>>>(W_out, WoutT, DIN, DD);

  // GEMM1: x_inner = x @ W_in + b_in  (f16 out)
  gemm_wmma<0><<<dim3(ROWS / 128, DIN / BN), 256, 0, stream>>>(
      xh, WinT, b_in, nullptr, xinH, nullptr, nullptr, nullptr, ROWS, DIN, DD);

  // basis projection + 16-wide LN
  k_wri<<<ROWS / 8, 256, 0, stream>>>(xinH, br, bi, lnw, lnb, vec);

  // sequential scan -> inv, final_state (written to tail of d_out)
  k_scan<<<1, 32, 0, stream>>>(vec, theta, omega, sigma, invb,
                               out + (size_t)ROWS * DD);

  // GEMM2 fused with silu gate and inv@W_sp epilogue -> h16
  gemm_wmma<2><<<dim3(ROWS / 128, DIN / BN), 256, 0, stream>>>(
      xinH, WgT, b_g, nullptr, h16, invb, W_sp, b_sp, ROWS, DIN, DIN);

  // GEMM3: pre = h @ W_out + b_out  (f32 out)
  gemm_wmma<1><<<dim3(ROWS / 128, DD / BN), 256, 0, stream>>>(
      h16, WoutT, b_out, pre, nullptr, nullptr, nullptr, nullptr, ROWS, DD, DIN);

  // final residual LayerNorm -> d_out
  k_final_ln<<<ROWS, 256, 0, stream>>>(x, pre, nw, nb, out);
}
